// Qwen2MLP_ND_56959856279853
// MI455X (gfx1250) — matrix operand order repaired and checked by reading, not compiled, not run on hardware
//
#include <hip/hip_runtime.h>
#include <hip/hip_bf16.h>

// ---------------------------------------------------------------------------
// Qwen2 SwiGLU MLP + neuron impacts for MI455X (gfx1250, wave32, WMMA).
// Compute-bound (~555 GFLOP vs ~15us of HBM traffic at 23.3 TB/s), so all
// three GEMMs run on v_wmma_f32_16x16x32_bf16 with fp32 accumulation.
// bf16 weights (~67MB) are L2-resident (192MB L2), so per-lane global
// fragment loads hit WGP$/L2 after the first block row. No explicit
// prefetch: relies on the compiler's own software pipelining
// (2x unroll, clause-grouped b128 loads, partial s_wait_loadcnt)
// ---------------------------------------------------------------------------

typedef __attribute__((ext_vector_type(16))) __bf16 v16bf;
typedef __attribute__((ext_vector_type(8)))  float  v8f;

#define B_DIM 4
#define S_DIM 2048
#define H_DIM 2048
#define I_DIM 5504
#define M_DIM (B_DIM * S_DIM)   /* 8192 tokens */

union Frag { v16bf v; uint4 q[2]; };   // 16 x bf16 = 32B = 8 VGPRs

__device__ __forceinline__ unsigned short f2bf(float f) {
    union { float f; unsigned u; } c; c.f = f;
    unsigned u = c.u + 0x7FFFu + ((c.u >> 16) & 1u);   // round-nearest-even
    return (unsigned short)(u >> 16);
}

// ---------------------------------------------------------------- converts
__global__ void k_cvt_bf16(const float* __restrict__ src,
                           unsigned short* __restrict__ dst, int n4) {
    int t = blockIdx.x * blockDim.x + threadIdx.x;
    if (t >= n4) return;
    float4 f = reinterpret_cast<const float4*>(src)[t];
    ushort4 o;
    o.x = f2bf(f.x); o.y = f2bf(f.y); o.z = f2bf(f.z); o.w = f2bf(f.w);
    reinterpret_cast<ushort4*>(dst)[t] = o;
}

// row_sq[i] = sum_h w_up[i,h]^2   (fp32 source for accuracy). 1 wave / row.
__global__ void k_rowsq(const float* __restrict__ wu, float* __restrict__ rs) {
    int wave = threadIdx.x >> 5, lane = threadIdx.x & 31;
    int row  = blockIdx.x * 8 + wave;
    const float* p = wu + (size_t)row * H_DIM;
    float s = 0.f;
    for (int k = lane; k < H_DIM; k += 32) { float v = p[k]; s += v * v; }
    #pragma unroll
    for (int off = 16; off > 0; off >>= 1) s += __shfl_xor(s, off, 32);
    if (lane == 0) rs[row] = s;
}

__global__ void k_zero(float* p, int n) {
    int t = blockIdx.x * blockDim.x + threadIdx.x;
    if (t < n) p[t] = 0.f;
}

// ------------------------------------------------------- GEMM1: gate + up
// Block tile 128(M) x 128(N); 8 waves as 2(M) x 4(N); wave tile 64 x 32 =
// 4x2 wmma tiles, two weight matrices -> 16 wmma per K-step of 32.
// Epilogue: inter = silu(gate)*up -> bf16 store + column sum of inter^2.
__global__ __launch_bounds__(256)
void k_gemm_gateup(const unsigned short* __restrict__ xb,
                   const unsigned short* __restrict__ wg,
                   const unsigned short* __restrict__ wu,
                   unsigned short* __restrict__ inter,
                   float* __restrict__ s_acc) {
    const int lane = threadIdx.x & 31;
    const int wave = threadIdx.x >> 5;
    const int half = lane >> 4;          // K-half selector (ISA A/B layouts)
    const int l16  = lane & 15;
    const int wm   = wave & 1;           // 2 wave rows (M)
    const int wn   = wave >> 1;          // 4 wave cols (N)
    const int mBase = blockIdx.y * 128 + wm * 64;
    const int nBase = blockIdx.x * 128 + wn * 32;

    // A lane rows (M = l16 within each 16-row tile)
    const unsigned short* aPtr[4];
    #pragma unroll
    for (int mt = 0; mt < 4; ++mt)
        aPtr[mt] = xb + (size_t)(mBase + mt * 16 + l16) * H_DIM + half * 8;

    // B lane rows (weight row n == output column), K = 8h..8h+7 and 16+8h.. per lane
    int nCol[2];
    const unsigned short *gPtr[2], *uPtr[2];
    #pragma unroll
    for (int nt = 0; nt < 2; ++nt) {
        nCol[nt] = nBase + nt * 16 + l16;
        gPtr[nt] = wg + (size_t)nCol[nt] * H_DIM + half * 8;
        uPtr[nt] = wu + (size_t)nCol[nt] * H_DIM + half * 8;
    }

    v8f accg[4][2], accu[4][2];
    #pragma unroll
    for (int mt = 0; mt < 4; ++mt)
        #pragma unroll
        for (int nt = 0; nt < 2; ++nt) { accg[mt][nt] = {}; accu[mt][nt] = {}; }

    for (int kt = 0; kt < H_DIM; kt += 32) {
        Frag a[4], bg[2], bu[2];
        #pragma unroll
        for (int mt = 0; mt < 4; ++mt) {
            const unsigned short* p = aPtr[mt] + kt;
            a[mt].q[0] = *reinterpret_cast<const uint4*>(p);       // K = 8h..8h+7
            a[mt].q[1] = *reinterpret_cast<const uint4*>(p + 16);  // K = 16+8h..
        }
        #pragma unroll
        for (int nt = 0; nt < 2; ++nt) {
            const unsigned short* pg = gPtr[nt] + kt;
            bg[nt].q[0] = *reinterpret_cast<const uint4*>(pg);
            bg[nt].q[1] = *reinterpret_cast<const uint4*>(pg + 16);
            const unsigned short* pu = uPtr[nt] + kt;
            bu[nt].q[0] = *reinterpret_cast<const uint4*>(pu);
            bu[nt].q[1] = *reinterpret_cast<const uint4*>(pu + 16);
        }
        #pragma unroll
        for (int mt = 0; mt < 4; ++mt)
            #pragma unroll
            for (int nt = 0; nt < 2; ++nt) {
                accg[mt][nt] = __builtin_amdgcn_wmma_f32_16x16x32_bf16(
                    false, a[mt].v, false, bg[nt].v, (short)0, accg[mt][nt], false, false);
                accu[mt][nt] = __builtin_amdgcn_wmma_f32_16x16x32_bf16(
                    false, a[mt].v, false, bu[nt].v, (short)0, accu[mt][nt], false, false);
            }
    }

    // Epilogue. Whole 128-row block lies in one batch (2048 % 128 == 0).
    const int b = blockIdx.y >> 4;   // (blockIdx.y*128)/2048
    #pragma unroll
    for (int nt = 0; nt < 2; ++nt) {
        const int col = nCol[nt];
        float colsum = 0.f;
        #pragma unroll
        for (int mt = 0; mt < 4; ++mt) {
            const int mtb = mBase + mt * 16;
            #pragma unroll
            for (int v = 0; v < 8; ++v) {
                float g   = accg[mt][nt][v];
                float u   = accu[mt][nt][v];
                float val = g * (1.f / (1.f + __expf(-g))) * u;   // silu(g)*u
                const int m = mtb + v + 8 * half;                 // C layout
                inter[(size_t)m * I_DIM + col] = f2bf(val);
                colsum += val * val;
            }
        }
        colsum += __shfl_xor(colsum, 16, 32);   // lanes l and l^16 share col
        if (half == 0)
            __hip_atomic_fetch_add(&s_acc[b * I_DIM + col], colsum,
                                   __ATOMIC_RELAXED, __HIP_MEMORY_SCOPE_AGENT);
    }
}

// --------------------------------------------------------- GEMM2: down-proj
// out[m,h] = sum_i inter[m,i] * w_down[h,i];  A=[M,K=I] bf16, B=[N=H,K] bf16.
__global__ __launch_bounds__(256)
void k_gemm_down(const unsigned short* __restrict__ inter,
                 const unsigned short* __restrict__ wd,
                 float* __restrict__ out) {
    const int lane = threadIdx.x & 31;
    const int wave = threadIdx.x >> 5;
    const int half = lane >> 4;
    const int l16  = lane & 15;
    const int wm   = wave & 1;
    const int wn   = wave >> 1;
    const int mBase = blockIdx.y * 128 + wm * 64;
    const int nBase = blockIdx.x * 128 + wn * 32;

    const unsigned short* aPtr[4];
    #pragma unroll
    for (int mt = 0; mt < 4; ++mt)
        aPtr[mt] = inter + (size_t)(mBase + mt * 16 + l16) * I_DIM + half * 8;

    int nCol[2];
    const unsigned short* dPtr[2];
    #pragma unroll
    for (int nt = 0; nt < 2; ++nt) {
        nCol[nt] = nBase + nt * 16 + l16;
        dPtr[nt] = wd + (size_t)nCol[nt] * I_DIM + half * 8;
    }

    v8f acc[4][2];
    #pragma unroll
    for (int mt = 0; mt < 4; ++mt)
        #pragma unroll
        for (int nt = 0; nt < 2; ++nt) acc[mt][nt] = {};

    for (int kt = 0; kt < I_DIM; kt += 32) {   // 5504 % 32 == 0
        Frag a[4], bd[2];
        #pragma unroll
        for (int mt = 0; mt < 4; ++mt) {
            const unsigned short* p = aPtr[mt] + kt;
            a[mt].q[0] = *reinterpret_cast<const uint4*>(p);
            a[mt].q[1] = *reinterpret_cast<const uint4*>(p + 16);
        }
        #pragma unroll
        for (int nt = 0; nt < 2; ++nt) {
            const unsigned short* pd = dPtr[nt] + kt;
            bd[nt].q[0] = *reinterpret_cast<const uint4*>(pd);
            bd[nt].q[1] = *reinterpret_cast<const uint4*>(pd + 16);
        }
        #pragma unroll
        for (int mt = 0; mt < 4; ++mt)
            #pragma unroll
            for (int nt = 0; nt < 2; ++nt)
                acc[mt][nt] = __builtin_amdgcn_wmma_f32_16x16x32_bf16(
                    false, a[mt].v, false, bd[nt].v, (short)0, acc[mt][nt], false, false);
    }

    #pragma unroll
    for (int nt = 0; nt < 2; ++nt) {
        const int col = nCol[nt];
        #pragma unroll
        for (int mt = 0; mt < 4; ++mt) {
            const int mtb = mBase + mt * 16;
            #pragma unroll
            for (int v = 0; v < 8; ++v)
                out[(size_t)(mtb + v + 8 * half) * H_DIM + col] = acc[mt][nt][v];
        }
    }
}

// ------------------------------------------------------------------ impacts
__global__ void k_impacts(const float* __restrict__ s_acc,
                          const float* __restrict__ rs,
                          float* __restrict__ impacts) {
    int t = blockIdx.x * blockDim.x + threadIdx.x;
    if (t < B_DIM * I_DIM) {
        int i = t % I_DIM;
        impacts[t] = sqrtf(s_acc[t] * rs[i]);
    }
}

// ------------------------------------------------------------------ launch
extern "C" void kernel_launch(void* const* d_in, const int* in_sizes, int n_in,
                              void* d_out, int out_size, void* d_ws, size_t ws_size,
                              hipStream_t stream) {
    (void)in_sizes; (void)n_in; (void)out_size; (void)ws_size;
    const float* hidden = (const float*)d_in[0];
    const float* w_gate = (const float*)d_in[1];
    const float* w_up   = (const float*)d_in[2];
    const float* w_down = (const float*)d_in[3];
    float* out = (float*)d_out;

    char* ws = (char*)d_ws;
    size_t off = 0;
    auto take = [&](size_t bytes) {
        void* p = ws + off;
        off += (bytes + 255) & ~(size_t)255;
        return p;
    };
    unsigned short* xb    = (unsigned short*)take((size_t)M_DIM * H_DIM * 2); // 32MB
    unsigned short* wgb   = (unsigned short*)take((size_t)I_DIM * H_DIM * 2); // 21.5MB
    unsigned short* wub   = (unsigned short*)take((size_t)I_DIM * H_DIM * 2);
    unsigned short* wdb   = (unsigned short*)take((size_t)H_DIM * I_DIM * 2);
    unsigned short* interb= (unsigned short*)take((size_t)M_DIM * I_DIM * 2); // 86MB
    float* rs    = (float*)take((size_t)I_DIM * 4);
    float* s_acc = (float*)take((size_t)B_DIM * I_DIM * 4);

    const int nx4 = M_DIM * H_DIM / 4;
    const int nw4 = I_DIM * H_DIM / 4;
    k_cvt_bf16<<<(nx4 + 255) / 256, 256, 0, stream>>>(hidden, xb, nx4);
    k_cvt_bf16<<<(nw4 + 255) / 256, 256, 0, stream>>>(w_gate, wgb, nw4);
    k_cvt_bf16<<<(nw4 + 255) / 256, 256, 0, stream>>>(w_up,   wub, nw4);
    k_cvt_bf16<<<(nw4 + 255) / 256, 256, 0, stream>>>(w_down, wdb, nw4);
    k_rowsq<<<I_DIM / 8, 256, 0, stream>>>(w_up, rs);
    k_zero<<<(B_DIM * I_DIM + 255) / 256, 256, 0, stream>>>(s_acc, B_DIM * I_DIM);

    // GEMM1: grid 43 x 64 (I/128 x M/128)
    k_gemm_gateup<<<dim3(I_DIM / 128, M_DIM / 128), 256, 0, stream>>>(
        xb, wgb, wub, interb, s_acc);

    k_impacts<<<(B_DIM * I_DIM + 255) / 256, 256, 0, stream>>>(
        s_acc, rs, out + (size_t)M_DIM * H_DIM);

    // GEMM2: grid 16 x 64 (H/128 x M/128)
    k_gemm_down<<<dim3(H_DIM / 128, M_DIM / 128), 256, 0, stream>>>(
        interb, wdb, out);
}
